// Llama4TextExperts_38036230373627
// MI455X (gfx1250) — compile-verified, matrix operand order checked by reading, not run on hardware
//
#include <hip/hip_runtime.h>
#include <hip/hip_bf16.h>

// Llama4TextExperts grouped MoE FFN for MI455X (gfx1250, wave32, WMMA).
// GEMM1+SwiGLU fused -> bf16 "gated" in workspace (64MB), then GEMM2 -> fp32 out.
// bf16 V_WMMA_F32_16X16X32_BF16 with f32 accumulation; double-buffered LDS
// (32KB of the 320KB WGP LDS) hides global-load latency behind the WMMA burst.
// Kernel 2's A tile uses GLOBAL_LOAD_ASYNC_TO_LDS_B128 (ASYNCcnt path).

#define NE 8
#define HID 2048
#define IDIM 4096
#define TOK 1024
#define TWO_I (2 * IDIM)

typedef __bf16 v16bf __attribute__((ext_vector_type(16)));
typedef float  v8f   __attribute__((ext_vector_type(8)));

struct U4  { unsigned x, y, z, w; };
struct B32 { U4 a, b; };  // 32 bytes == v16bf

#if __has_builtin(__builtin_amdgcn_global_load_async_to_lds_b128)
#define HAVE_ASYNC_LDS 1
typedef int v4i_ __attribute__((vector_size(16)));
typedef __attribute__((address_space(1))) v4i_* g_v4i_ptr;
typedef __attribute__((address_space(3))) v4i_* l_v4i_ptr;
#endif

// pack two floats into one dword of bf16 pairs {lo=a, hi=b}
static __device__ __forceinline__ unsigned pk(float a, float b) {
    __bf16 x = (__bf16)a, y = (__bf16)b;
    unsigned short ux = __builtin_bit_cast(unsigned short, x);
    unsigned short uy = __builtin_bit_cast(unsigned short, y);
    return ((unsigned)uy << 16) | (unsigned)ux;
}

// LDS tile layout: row r = 16 dwords, dword kp holds bf16 pair (K=2kp, 2kp+1).
// WMMA 16x32 bf16 A/B fragment for lane (half, r):
//   VGPRs 0-3: bytes r*64 + half*16 ; VGPRs 4-7: bytes r*64 + 32 + half*16
static __device__ __forceinline__ v16bf load_frag(const unsigned* lds, int row, int half) {
    const U4* p = (const U4*)(lds + row * 16);
    B32 t;
    t.a = p[half];
    t.b = p[2 + half];
    return __builtin_bit_cast(v16bf, t);
}

// 16B global -> LDS copy; async (ASYNCcnt-tracked) path when available.
static __device__ __forceinline__ void copy16_g2l(const void* g, void* l) {
#ifdef HAVE_ASYNC_LDS
    __builtin_amdgcn_global_load_async_to_lds_b128(
        (g_v4i_ptr)(uintptr_t)g,
        (l_v4i_ptr)(unsigned)(uintptr_t)l,
        0, 0);
#else
    *(U4*)l = *(const U4*)g;
#endif
}

static __device__ __forceinline__ void wait_async_done() {
#ifdef HAVE_ASYNC_LDS
#if __has_builtin(__builtin_amdgcn_s_wait_asynccnt)
    __builtin_amdgcn_s_wait_asynccnt(0);
#else
    asm volatile("s_wait_asynccnt 0x0" ::: "memory");
#endif
#endif
}

// ---------------- Kernel 1: gate_up GEMM + SwiGLU, output bf16 gated ----------------
// C tile: BM=128 rows x (64 gate cols | 64 up cols). 8 waves, each 16 rows x 8 tiles.
__global__ __launch_bounds__(256)
void moe_gateup_swiglu(const float* __restrict__ h,
                       const float* __restrict__ gup,
                       unsigned short* __restrict__ gated) {
    __shared__ unsigned ldsA[2][128 * 16];  // 2 x 8KB ping-pong
    __shared__ unsigned ldsB[2][128 * 16];

    const int e   = blockIdx.z;
    const int m0  = blockIdx.y * 128;
    const int n0  = blockIdx.x * 64;   // gate column base within IDIM
    const int tid  = threadIdx.x;
    const int wave = tid >> 5;
    const int lane = tid & 31;
    const int half = lane >> 4;
    const int ln   = lane & 15;

    const float* A = h + ((size_t)e * TOK + m0) * HID;
    const float* B = gup + (size_t)e * HID * TWO_I;

    const int kfA = tid & 7;    // float4 slot along K for A (8 per row)
    const int r0A = tid >> 3;   // 0..31
    const int kpB = tid & 15;   // K-pair 0..15 for B
    const int ngB = tid >> 4;   // 0..15

    auto stageA = [&](unsigned* dA, int k0) {
#pragma unroll
        for (int it = 0; it < 4; ++it) {
            const int m = r0A + 32 * it;
            const float4 v = *(const float4*)(A + (size_t)m * HID + k0 + kfA * 4);
            unsigned* d = dA + m * 16 + kfA * 2;
            d[0] = pk(v.x, v.y);
            d[1] = pk(v.z, v.w);
        }
    };
    auto stageB = [&](unsigned* dB, int k0) {
#pragma unroll
        for (int it = 0; it < 2; ++it) {
            const int n   = (ngB + 16 * it) * 4;                       // 0..127
            const int col = (n < 64) ? (n0 + n) : (IDIM + n0 + (n - 64));
            const float* src = B + (size_t)(k0 + 2 * kpB) * TWO_I + col;
            const float4 v0 = *(const float4*)src;
            const float4 v1 = *(const float4*)(src + TWO_I);
            unsigned* d = dB + n * 16 + kpB;
            d[0]  = pk(v0.x, v1.x);
            d[16] = pk(v0.y, v1.y);
            d[32] = pk(v0.z, v1.z);
            d[48] = pk(v0.w, v1.w);
        }
    };

    v8f zero = {};
    v8f acc[8];
#pragma unroll
    for (int i = 0; i < 8; ++i) acc[i] = zero;

    stageA(ldsA[0], 0);
    stageB(ldsB[0], 0);
    __syncthreads();

    int buf = 0;
    for (int k0 = 0; k0 < HID; k0 += 32) {
        const int kn = k0 + 32;
        if (kn < HID) {                        // stage next slice into other buffer
            stageA(ldsA[buf ^ 1], kn);
            stageB(ldsB[buf ^ 1], kn);
            __builtin_prefetch(A + (size_t)r0A * HID + kn + 32, 0, 1);
            __builtin_prefetch(B + (size_t)(kn + 32 + 2 * kpB) * TWO_I + n0, 0, 1);
        }
        // compute on current buffer: preload all B frags, then 8-WMMA burst
        const v16bf af = load_frag(ldsA[buf], wave * 16 + ln, half);
        v16bf bfr[8];
#pragma unroll
        for (int nt = 0; nt < 8; ++nt) bfr[nt] = load_frag(ldsB[buf], nt * 16 + ln, half);
#pragma unroll
        for (int nt = 0; nt < 8; ++nt)
            acc[nt] = __builtin_amdgcn_wmma_f32_16x16x32_bf16(
                false, af, false, bfr[nt], (short)0, acc[nt], false, false);
        __syncthreads();
        buf ^= 1;
    }

    // ---- SwiGLU epilogue: tiles 0-3 = gate, 4-7 = up (same lane/VGPR -> same (m,c)) ----
    const size_t orow = (size_t)e * TOK + m0 + wave * 16;
#pragma unroll
    for (int nt = 0; nt < 4; ++nt) {
        const int i = n0 + nt * 16 + ln;   // C layout: N = lane%16
#pragma unroll
        for (int r = 0; r < 8; ++r) {
            const int m = r + 8 * half;    // C layout: M = vgpr + 8*(lane/16)
            const float g = acc[nt][r];
            const float u = acc[nt + 4][r];
            const float val = u * (g / (1.0f + __expf(-g)));   // up * silu(gate)
            __bf16 bv = (__bf16)val;
            gated[(orow + m) * IDIM + i] = __builtin_bit_cast(unsigned short, bv);
        }
    }
}

// ---------------- Kernel 2: down GEMM (bf16 gated x bf16(down_proj)) -> fp32 out ----------------
__global__ __launch_bounds__(256)
void moe_down(const unsigned short* __restrict__ gated,
              const float* __restrict__ dwn,
              float* __restrict__ out) {
    __shared__ unsigned ldsA[2][128 * 16];
    __shared__ unsigned ldsB[2][128 * 16];

    const int e   = blockIdx.z;
    const int m0  = blockIdx.y * 128;
    const int n0  = blockIdx.x * 128;
    const int tid  = threadIdx.x;
    const int wave = tid >> 5;
    const int lane = tid & 31;
    const int half = lane >> 4;
    const int ln   = lane & 15;

    // dword view of gated rows: row stride = IDIM/2 dwords, dword kp = pair(2kp,2kp+1)
    const unsigned* Ad = (const unsigned*)gated + ((size_t)e * TOK + m0) * (IDIM / 2);
    const float*    Bd = dwn + (size_t)e * IDIM * HID;

    const int cA  = tid & 3;    // 16B chunk within row (4 per 32-K row)
    const int rA  = tid >> 2;   // 0..63
    const int kpB = tid & 15;
    const int ngB = tid >> 4;

    // A tile is already bf16 in fragment-pair order -> async global->LDS copy
    auto stageA = [&](unsigned* dA, int k0) {
#pragma unroll
        for (int it = 0; it < 2; ++it) {
            const int m = rA + 64 * it;
            copy16_g2l(Ad + (size_t)m * (IDIM / 2) + k0 / 2 + cA * 4,
                       dA + m * 16 + cA * 4);
        }
    };
    auto stageB = [&](unsigned* dB, int k0) {
#pragma unroll
        for (int it = 0; it < 2; ++it) {
            const int n = (ngB + 16 * it) * 4;
            const float* src = Bd + (size_t)(k0 + 2 * kpB) * HID + n0 + n;
            const float4 v0 = *(const float4*)src;
            const float4 v1 = *(const float4*)(src + HID);
            unsigned* d = dB + n * 16 + kpB;
            d[0]  = pk(v0.x, v1.x);
            d[16] = pk(v0.y, v1.y);
            d[32] = pk(v0.z, v1.z);
            d[48] = pk(v0.w, v1.w);
        }
    };

    v8f zero = {};
    v8f acc[8];
#pragma unroll
    for (int i = 0; i < 8; ++i) acc[i] = zero;

    stageA(ldsA[0], 0);
    stageB(ldsB[0], 0);
    wait_async_done();
    __syncthreads();

    int buf = 0;
    for (int k0 = 0; k0 < IDIM; k0 += 32) {
        const int kn = k0 + 32;
        if (kn < IDIM) {
            stageA(ldsA[buf ^ 1], kn);
            stageB(ldsB[buf ^ 1], kn);
            __builtin_prefetch(Bd + (size_t)(kn + 32 + 2 * kpB) * HID + n0, 0, 1);
        }
        const v16bf af = load_frag(ldsA[buf], wave * 16 + ln, half);
        v16bf bfr[8];
#pragma unroll
        for (int nt = 0; nt < 8; ++nt) bfr[nt] = load_frag(ldsB[buf], nt * 16 + ln, half);
#pragma unroll
        for (int nt = 0; nt < 8; ++nt)
            acc[nt] = __builtin_amdgcn_wmma_f32_16x16x32_bf16(
                false, af, false, bfr[nt], (short)0, acc[nt], false, false);
        wait_async_done();   // this wave's async copies for buf^1 landed
        __syncthreads();
        buf ^= 1;
    }

    const size_t orow = (size_t)e * TOK + m0 + wave * 16;
#pragma unroll
    for (int nt = 0; nt < 8; ++nt) {
        const int c = n0 + nt * 16 + ln;
#pragma unroll
        for (int r = 0; r < 8; ++r) {
            const int m = r + 8 * half;
            out[(orow + m) * HID + c] = acc[nt][r];
        }
    }
}

extern "C" void kernel_launch(void* const* d_in, const int* in_sizes, int n_in,
                              void* d_out, int out_size, void* d_ws, size_t ws_size,
                              hipStream_t stream) {
    const float* h   = (const float*)d_in[0];  // [E*T, H] fp32
    const float* gup = (const float*)d_in[1];  // [E, H, 2I] fp32
    const float* dwn = (const float*)d_in[2];  // [E, I, H] fp32
    float* out = (float*)d_out;                // [E*T, H] fp32
    // workspace: gated activation, bf16 [E, T, I] = 64 MB
    unsigned short* gated = (unsigned short*)d_ws;

    dim3 blk(256);
    // grid1: 64 n-tiles (of 64 gate+64 up cols) x 8 m-tiles x 8 experts
    moe_gateup_swiglu<<<dim3(IDIM / 64, TOK / 128, NE), blk, 0, stream>>>(h, gup, gated);
    // grid2: 16 n-tiles x 8 m-tiles x 8 experts
    moe_down<<<dim3(HID / 128, TOK / 128, NE), blk, 0, stream>>>(gated, dwn, out);
}
